// Exchangable_25503515803842
// MI455X (gfx1250) — matrix operand order repaired and checked by reading, not compiled, not run on hardware
//
#include <hip/hip_runtime.h>

typedef __attribute__((ext_vector_type(16))) __bf16 v16bf;
typedef __attribute__((ext_vector_type(8)))  float  v8f;
typedef __attribute__((ext_vector_type(4)))  float  f4;

static __device__ inline v8f wmma_bf16(v16bf a, v16bf b, v8f c) {
    // (neg_a, A, neg_b, B, c_mod, C, reuse_a, reuse_b)
    return __builtin_amdgcn_wmma_f32_16x16x32_bf16(false, a, false, b, (short)0, c, false, false);
}

// Build bf16 hi/lo fragments from two contiguous 8-float runs (p[0..7], q[0..7]).
static __device__ inline void make_frag(const float* __restrict__ p,
                                        const float* __restrict__ q,
                                        float scale, v16bf& hi, v16bf& lo) {
    f4 x0 = ((const f4*)p)[0];
    f4 x1 = ((const f4*)p)[1];
    f4 x2 = ((const f4*)q)[0];
    f4 x3 = ((const f4*)q)[1];
    float a[16] = {x0.x, x0.y, x0.z, x0.w, x1.x, x1.y, x1.z, x1.w,
                   x2.x, x2.y, x2.z, x2.w, x3.x, x3.y, x3.z, x3.w};
#pragma unroll
    for (int e = 0; e < 16; ++e) {
        float v = a[e] * scale;
        __bf16 h = (__bf16)v;         // RNE truncate: high part
        hi[e] = h;
        lo[e] = (__bf16)(v - (float)h);  // residual: low part
    }
}

// ---------------- Kernel 1: zero the workspace (pool tables + mean acc) ----------------
__global__ void exch_zero_kernel(float* __restrict__ ws,
                                 const int* __restrict__ pnr,
                                 const int* __restrict__ pnc) {
    long long total  = ((long long)(*pnr) + (long long)(*pnc)) * 32 + 32;
    long long stride = (long long)gridDim.x * blockDim.x;
    for (long long i = (long long)blockIdx.x * blockDim.x + threadIdx.x; i < total; i += stride)
        ws[i] = 0.0f;
}

// ---------------- Kernel 2: segment-sum scatter (L2-resident atomics) + mean ----------------
__global__ void exch_scatter_kernel(const float* __restrict__ values,
                                    const int* __restrict__ indices,
                                    const int* __restrict__ pnr,
                                    const int* __restrict__ pnc,
                                    float* __restrict__ ws,
                                    long long nnz) {
    const long long nr = *pnr, nc = *pnc;
    float* row_sums = ws;                       // [nr, 32]  (pooled over cols)
    float* col_sums = ws + nr * 32;             // [nc, 32]  (pooled over rows)
    float* mean_acc = ws + (nr + nc) * 32;      // [32]

    __shared__ float smean[32];
    if (threadIdx.x < 32) smean[threadIdx.x] = 0.0f;
    __syncthreads();

    const long long total  = nnz * 32;
    const long long stride = (long long)gridDim.x * blockDim.x;  // multiple of 32 -> d fixed per thread
    float acc = 0.0f;
    for (long long idx = (long long)blockIdx.x * blockDim.x + threadIdx.x; idx < total; idx += stride) {
        long long i = idx >> 5;
        int d = (int)(idx & 31);
        float v = values[idx];
        int r = indices[i];
        int c = indices[nnz + i];
        atomicAdd(&row_sums[(long long)r * 32 + d], v);
        atomicAdd(&col_sums[(long long)c * 32 + d], v);
        acc += v;
    }
    atomicAdd(&smean[threadIdx.x & 31], acc);
    __syncthreads();
    if (threadIdx.x < 32) atomicAdd(&mean_acc[threadIdx.x], smean[threadIdx.x]);
}

// ---------------- Kernel 3: fused gather + WMMA GEMM + bias + leaky ReLU ----------------
__global__ __launch_bounds__(256) void exch_gemm_kernel(
        const float* __restrict__ values, const int* __restrict__ indices,
        const float* __restrict__ W, const float* __restrict__ bias,
        const int* __restrict__ pnr, const int* __restrict__ pnc,
        const float* __restrict__ ws, float* __restrict__ out,
        long long nnz, float inv_nnz) {
    const int lane = threadIdx.x & 31;
    const int half = lane >> 4;     // 0: lanes 0-15, 1: lanes 16-31
    const int nsub = lane & 15;
    const int kb   = half * 8;      // A-fragment K base within a 32-wide segment

    const long long nr = *pnr, nc = *pnc;
    const float* row_sums = ws;
    const float* col_sums = ws + nr * 32;
    const float* mean_acc = ws + (nr + nc) * 32;

    // ---- Preload B = W^T fragments (bf16 hi/lo), resident across the group loop ----
    // B(32x16) layout: lane holds N = nsub (+16u), K = half*8 + e (e<8), 16 + half*8 + (e-8) (e>=8).
    v16bf bhi[4][2], blo[4][2];
#pragma unroll
    for (int u = 0; u < 2; ++u) {
#pragma unroll
        for (int t = 0; t < 4; ++t) {
            const float* wrow = W + (long long)(u * 16 + nsub) * 128 + t * 32 + half * 8;
            make_frag(wrow, wrow + 16, 1.0f, bhi[t][u], blo[t][u]);
        }
    }
    const float bias0 = bias[nsub];
    const float bias1 = bias[16 + nsub];

    const long long groups = (nnz + 15) >> 4;
    const long long wave   = (long long)blockIdx.x * (blockDim.x >> 5) + (threadIdx.x >> 5);
    const long long nwaves = (long long)gridDim.x * (blockDim.x >> 5);

    for (long long g = wave; g < groups; g += nwaves) {
        long long r  = g * 16 + nsub;
        long long rl = (r < nnz) ? r : (nnz - 1);    // clamp tail rows (dupes; stores are guarded)
        int ri = indices[rl];            // row index  -> col_pool = row_sums[ri]
        int ci = indices[nnz + rl];      // col index  -> row_pool = col_sums[ci]

        // Prefetch next group's values row into cache.
        long long rp = g + nwaves < groups ? (g + nwaves) * 16 + nsub : rl;
        if (rp >= nnz) rp = nnz - 1;
        __builtin_prefetch(values + rp * 32, 0, 1);

        const float* sp[4];
        sp[0] = values   + rl * 32;              // segment 0: values
        sp[1] = col_sums + (long long)ci * 32;   // segment 1: row_pool
        sp[2] = row_sums + (long long)ri * 32;   // segment 2: col_pool
        sp[3] = mean_acc;                        // segment 3: both (mean = acc * inv_nnz)
        const float sc[4] = {1.0f, 1.0f, 1.0f, inv_nnz};

        v8f c0 = {};  // output cols 0..15
        v8f c1 = {};  // output cols 16..31
#pragma unroll
        for (int t = 0; t < 4; ++t) {
            v16bf ahi, alo;
            make_frag(sp[t] + kb, sp[t] + 16 + kb, sc[t], ahi, alo);
            // bf16x3 error-compensated product, fp32 accumulate
            c0 = wmma_bf16(ahi, bhi[t][0], c0);
            c0 = wmma_bf16(alo, bhi[t][0], c0);
            c0 = wmma_bf16(ahi, blo[t][0], c0);
            c1 = wmma_bf16(ahi, bhi[t][1], c1);
            c1 = wmma_bf16(alo, bhi[t][1], c1);
            c1 = wmma_bf16(ahi, blo[t][1], c1);
        }

        // Epilogue: bias + leaky ReLU, store. D element v -> row g*16 + v + half*8, col nsub (+16).
#pragma unroll
        for (int v = 0; v < 8; ++v) {
            long long m = g * 16 + v + half * 8;
            if (m < nnz) {
                float o0 = c0[v] + bias0;
                o0 = (o0 >= 0.0f) ? o0 : 0.01f * o0;
                float o1 = c1[v] + bias1;
                o1 = (o1 >= 0.0f) ? o1 : 0.01f * o1;
                out[m * 32 + nsub]      = o0;
                out[m * 32 + 16 + nsub] = o1;
            }
        }
    }
}

extern "C" void kernel_launch(void* const* d_in, const int* in_sizes, int n_in,
                              void* d_out, int out_size, void* d_ws, size_t ws_size,
                              hipStream_t stream) {
    const float* values  = (const float*)d_in[0];   // [nnz, 32] f32
    const int*   indices = (const int*)d_in[1];     // [2, nnz]  i32
    const float* W       = (const float*)d_in[2];   // [32, 128] f32
    const float* bias    = (const float*)d_in[3];   // [32]      f32
    const int*   pnr     = (const int*)d_in[4];     // scalar n_rows (device)
    const int*   pnc     = (const int*)d_in[5];     // scalar n_cols (device)

    long long nnz = (long long)in_sizes[0] / 32;
    float inv_nnz = 1.0f / (float)nnz;

    float* ws  = (float*)d_ws;
    float* out = (float*)d_out;

    exch_zero_kernel<<<2048, 256, 0, stream>>>(ws, pnr, pnc);
    exch_scatter_kernel<<<4096, 256, 0, stream>>>(values, indices, pnr, pnc, ws, nnz);
    exch_gemm_kernel<<<2048, 256, 0, stream>>>(values, indices, W, bias, pnr, pnc,
                                               ws, out, nnz, inv_nnz);
}
